// GraphAttentionLayer_36292473651250
// MI455X (gfx1250) — matrix operand order repaired and checked by reading, not compiled, not run on hardware
//
#include <hip/hip_runtime.h>
#include <hip/hip_bf16.h>

#define NN 2048
#define FF 64
#define ALPHA 0.2f
#define NEG_BIG -9000000000000000.0f

typedef __attribute__((ext_vector_type(16))) _Float16 v16h;
typedef __attribute__((ext_vector_type(8)))  float    v8f;

// ---------------------------------------------------------------------------
// Kernel 1: x = input*(nw+1)+nb ; h = x@W ; layernorm ; s1,s2 dots.
// One wave per node (32 lanes x 2 features). W/a/gamma/beta in LDS.
// Outputs: h32 [B][N][64] f32 (residual), hT16 [B][64][N] f16 (WMMA B),
//          s1 [B*N], s2 [B*N].
// ---------------------------------------------------------------------------
__global__ __launch_bounds__(256) void gat_prep_kernel(
    const float* __restrict__ input, const float* __restrict__ W,
    const float* __restrict__ a, const float* __restrict__ nw,
    const float* __restrict__ nb, const float* __restrict__ gamma,
    const float* __restrict__ beta,
    float* __restrict__ h32, _Float16* __restrict__ hT16,
    float* __restrict__ s1o, float* __restrict__ s2o)
{
    __shared__ float Wl[64 * 64];
    __shared__ float a1[64], a2[64], gl[64], bl[64];
    __shared__ float xw[8][64];

    const int tid = threadIdx.x;
    for (int i = tid; i < 4096; i += 256) Wl[i] = W[i];
    if (tid < 64) { a1[tid] = a[tid]; a2[tid] = a[64 + tid];
                    gl[tid] = gamma[tid]; bl[tid] = beta[tid]; }

    const int wave = tid >> 5, lane = tid & 31;
    const long node = (long)blockIdx.x * 8 + wave;   // 0..16383
    const int b = (int)(node >> 11);
    const int n = (int)(node & 2047);

    const float scale = nw[n] + 1.0f;                // input*nw + nb + input
    const float bias  = nb[n];
    const float2 xv = ((const float2*)(input + node * 64))[lane];
    xw[wave][2 * lane]     = xv.x * scale + bias;
    xw[wave][2 * lane + 1] = xv.y * scale + bias;
    __syncthreads();

    const int f0 = lane, f1 = lane + 32;
    float acc0 = 0.f, acc1 = 0.f;
    #pragma unroll
    for (int k = 0; k < 64; ++k) {
        const float xk = xw[wave][k];
        acc0 += xk * Wl[k * 64 + f0];
        acc1 += xk * Wl[k * 64 + f1];
    }
    // layernorm over 64 features (wave32 reductions)
    float s = acc0 + acc1;
    #pragma unroll
    for (int m = 16; m >= 1; m >>= 1) s += __shfl_xor(s, m, 32);
    const float mu = s * (1.0f / 64.0f);
    const float d0 = acc0 - mu, d1 = acc1 - mu;
    float q = d0 * d0 + d1 * d1;
    #pragma unroll
    for (int m = 16; m >= 1; m >>= 1) q += __shfl_xor(q, m, 32);
    const float rsq = rsqrtf(q * (1.0f / 64.0f) + 1e-5f);
    const float h0 = d0 * rsq * gl[f0] + bl[f0];
    const float h1 = d1 * rsq * gl[f1] + bl[f1];

    float p1 = h0 * a1[f0] + h1 * a1[f1];
    float p2 = h0 * a2[f0] + h1 * a2[f1];
    #pragma unroll
    for (int m = 16; m >= 1; m >>= 1) {
        p1 += __shfl_xor(p1, m, 32);
        p2 += __shfl_xor(p2, m, 32);
    }

    h32[node * 64 + f0] = h0;
    h32[node * 64 + f1] = h1;
    hT16[((long)b * 64 + f0) * NN + n] = (_Float16)h0;
    hT16[((long)b * 64 + f1) * NN + n] = (_Float16)h1;
    if (lane == 0) { s1o[node] = p1; s2o[node] = p2; }
}

// ---------------------------------------------------------------------------
// Kernel 2: pack adj (N x N int32) into bitmask adjb [N][64] u32 (512 KB).
// ---------------------------------------------------------------------------
__global__ __launch_bounds__(256) void gat_adjbits_kernel(
    const int* __restrict__ adj, unsigned* __restrict__ adjb)
{
    const int idx = blockIdx.x * 256 + threadIdx.x;  // 0..131071 = i*64 + w
    const int4* p = (const int4*)(adj + (long)idx * 32);
    unsigned bits = 0u;
    #pragma unroll
    for (int q = 0; q < 8; ++q) {
        const int4 v = p[q];
        bits |= (v.x > 0 ? 1u : 0u) << (q * 4)
             |  (v.y > 0 ? 1u : 0u) << (q * 4 + 1)
             |  (v.z > 0 ? 1u : 0u) << (q * 4 + 2)
             |  (v.w > 0 ? 1u : 0u) << (q * 4 + 3);
    }
    adjb[idx] = bits;
}

// ---------------------------------------------------------------------------
// Kernel 3: row softmax max. leaky_relu is monotonic, so
//   max_j e_ij = lrelu(s1_i + max_{j in adj(i)} s2_j), or NEG_BIG if no nbrs.
// One wave per (b,i) row; lane = bit position in each adjb word.
// ---------------------------------------------------------------------------
__global__ __launch_bounds__(128) void gat_rowmax_kernel(
    const unsigned* __restrict__ adjb, const float* __restrict__ s1,
    const float* __restrict__ s2, float* __restrict__ M)
{
    const int wid  = blockIdx.x * 4 + (threadIdx.x >> 5); // 0..16383 = b*2048+i
    const int lane = threadIdx.x & 31;
    const int i = wid & 2047;
    const int b = wid >> 11;
    const unsigned* row = adjb + i * 64;
    const float* s2b = s2 + b * NN;

    float mx = -3.4e38f;
    unsigned any = 0u;
    for (int it = 0; it < 64; ++it) {
        const unsigned w = row[it];
        any |= w;
        const float sv = s2b[it * 32 + lane];
        if ((w >> lane) & 1u) mx = fmaxf(mx, sv);
    }
    #pragma unroll
    for (int m = 16; m >= 1; m >>= 1) mx = fmaxf(mx, __shfl_xor(mx, m, 32));
    float outv;
    if (any) { const float v = s1[wid] + mx; outv = v > 0.f ? v : ALPHA * v; }
    else     { outv = NEG_BIG; }
    if (lane == 0) M[wid] = outv;
}

// ---------------------------------------------------------------------------
// Kernel 4: fused flash-attention pass.
// Block = 128 threads (4 waves); each wave owns 16 query rows of one batch.
// Double-buffered LDS (single barrier/iter), all 8 B-tile ds_load_b128 pinned
// before the exp chain via sched_barrier so one dscnt wait covers all WMMAs,
// P built in the 16-bit A-matrix VGPR layout, 4x v_wmma_f32_16x16x32_f16 for
// att@h plus a 5th WMMA against an all-ones B for the softmax row-sum.
// Finalize: fast-rcp /l, +h residual, ELU.
// ---------------------------------------------------------------------------
__global__ __launch_bounds__(128) void gat_flash_kernel(
    const _Float16* __restrict__ hT, const float* __restrict__ h32,
    const float* __restrict__ s1, const float* __restrict__ s2,
    const float* __restrict__ M, const unsigned* __restrict__ adjb,
    float* __restrict__ out)
{
    // [buf][f][kk] packed f16 pairs; pad 20 u32 -> 16B-aligned b128 rows,
    // gcd(20,64)=4 spreads lanes over all banks.
    __shared__ __align__(16) unsigned ldsB[2][64 * 20];

    const int tid = threadIdx.x;
    const int wave = tid >> 5, lane = tid & 31;
    const int b  = blockIdx.x >> 5;                    // 32 blocks per batch
    const int i0 = (blockIdx.x & 31) * 64 + wave * 16; // this wave's row tile
    const int row16 = lane & 15, hi = lane >> 4;
    const int sel8 = hi * 8;
    const int i = i0 + row16;

    const float s1v  = s1[b * NN + i];
    const float mrow = M[b * NN + i];
    const float* s2b = s2 + b * NN;
    const unsigned* adjrow = adjb + i * 64;
    const uint4* hT4 = (const uint4*)(hT + (long)b * 64 * NN); // [f][256] uint4

    // cooperative-fill mapping: 256 uint4 per tile, 2 per thread
    const int fa = tid >> 2,          ka = tid & 3;          // f 0..31
    const int fb = (128 + tid) >> 2,  kb = tid & 3;          // f 32..63
    const uint4* gA = hT4 + fa * 256 + ka;   // + jt*4 per tile
    const uint4* gB = hT4 + fb * 256 + kb;

    // all-ones B for the row-sum WMMA
    v16h ones;
    #pragma unroll
    for (int k = 0; k < 16; ++k) ones[k] = (_Float16)1.0f;

    v8f acc0 = {}, acc1 = {}, acc2 = {}, acc3 = {}, accl = {};

    uint4 ra = gA[0], rb = gB[0];   // prefetch tile 0
    int p = 0;

    for (int jt = 0; jt < 64; ++jt) {
        // store prefetched tile into current buffer
        *(uint4*)&ldsB[p][fa * 20 + ka * 4] = ra;
        *(uint4*)&ldsB[p][fb * 20 + kb * 4] = rb;
        __syncthreads();   // only barrier: publishes buf p; double-buffering
                           // + in-order DS covers reuse of buf p^1

        // issue next tile's global loads (overlap with compute below; their
        // wait can slide to next iteration's ds_store)
        const int jn = (jt + 1) & 63;
        ra = gA[jn * 4];
        rb = gB[jn * 4];

        // issue ALL 8 B-tile b128 loads up front into distinct registers;
        // the sched_barrier pins them here so a single dscnt wait covers the
        // whole WMMA group and the exp chain runs under LDS latency.
        const unsigned* Bb = &ldsB[p][row16 * 20 + hi * 4];
        union { v16h h; uint4 q[2]; } Bt0, Bt1, Bt2, Bt3;
        Bt0.q[0] = *(const uint4*)(Bb);        Bt0.q[1] = *(const uint4*)(Bb + 8);
        Bt1.q[0] = *(const uint4*)(Bb + 320);  Bt1.q[1] = *(const uint4*)(Bb + 328);
        Bt2.q[0] = *(const uint4*)(Bb + 640);  Bt2.q[1] = *(const uint4*)(Bb + 648);
        Bt3.q[0] = *(const uint4*)(Bb + 960);  Bt3.q[1] = *(const uint4*)(Bb + 968);
        __builtin_amdgcn_sched_barrier(0);

        // --- P tile in A-matrix layout (lane row = lane&15) ---
        const unsigned w = adjrow[jt];
        const unsigned wa = w >> sel8;          // bits 0..7 -> K group 1
        const unsigned wb = w >> (16 + sel8);   // bits 0..7 -> K group 2
        const int j0 = jt * 32;
        const float4 sA0 = *(const float4*)(s2b + j0 + sel8);
        const float4 sA1 = *(const float4*)(s2b + j0 + sel8 + 4);
        const float4 sB0 = *(const float4*)(s2b + j0 + 16 + sel8);
        const float4 sB1 = *(const float4*)(s2b + j0 + 16 + sel8 + 4);
        const float sarr[16] = { sA0.x, sA0.y, sA0.z, sA0.w,
                                 sA1.x, sA1.y, sA1.z, sA1.w,
                                 sB0.x, sB0.y, sB0.z, sB0.w,
                                 sB1.x, sB1.y, sB1.z, sB1.w };
        v16h A;
        #pragma unroll
        for (int k = 0; k < 16; ++k) {
            const unsigned bit = (k < 8) ? ((wa >> k) & 1u) : ((wb >> (k - 8)) & 1u);
            const float v = s1v + sarr[k];
            const float lr = fmaxf(v, ALPHA * v);           // leaky_relu
            const float e  = bit ? lr : NEG_BIG;            // exact NEG_BIG when masked
            A[k] = (_Float16)__expf(e - mrow);              // ->0 masked; ->1 if m=NEG_BIG
        }

        acc0 = __builtin_amdgcn_wmma_f32_16x16x32_f16(false, A, false, Bt0.h, (short)0, acc0, false, false);
        acc1 = __builtin_amdgcn_wmma_f32_16x16x32_f16(false, A, false, Bt1.h, (short)0, acc1, false, false);
        acc2 = __builtin_amdgcn_wmma_f32_16x16x32_f16(false, A, false, Bt2.h, (short)0, acc2, false, false);
        acc3 = __builtin_amdgcn_wmma_f32_16x16x32_f16(false, A, false, Bt3.h, (short)0, acc3, false, false);
        // softmax denominator via matrix pipe: row sums land in C layout
        accl = __builtin_amdgcn_wmma_f32_16x16x32_f16(false, A, false, ones,  (short)0, accl, false, false);

        p ^= 1;
    }

    // --- finalize: fast-rcp row-sum scale (C layout), +h residual, ELU ---
    #pragma unroll
    for (int r = 0; r < 8; ++r) {
        const float inv = __builtin_amdgcn_rcpf(accl[r]);  // l >= 1 always
        const long base = ((long)(b * NN + i0 + r + hi * 8)) * 64;
        const float vals[4] = { acc0[r], acc1[r], acc2[r], acc3[r] };
        #pragma unroll
        for (int t = 0; t < 4; ++t) {
            const int f = t * 16 + row16;                  // C-layout col = lane&15
            const float v = vals[t] * inv + h32[base + f];
            out[base + f] = v > 0.f ? v : (__expf(v) - 1.0f);
        }
    }
}

// ---------------------------------------------------------------------------
extern "C" void kernel_launch(void* const* d_in, const int* in_sizes, int n_in,
                              void* d_out, int out_size, void* d_ws, size_t ws_size,
                              hipStream_t stream) {
    const float* input = (const float*)d_in[0];   // [8,2048,64]
    const int*   adj   = (const int*)  d_in[1];   // [2048,2048]
    const float* W     = (const float*)d_in[2];   // [64,64]
    const float* a     = (const float*)d_in[3];   // [128,1]
    const float* nw    = (const float*)d_in[4];   // [2048]
    const float* nb    = (const float*)d_in[5];   // [2048]
    const float* gam   = (const float*)d_in[6];   // [64]
    const float* bet   = (const float*)d_in[7];   // [64]
    float* out = (float*)d_out;                   // [8,2048,64]

    char* ws = (char*)d_ws;
    float*     h32  = (float*)ws;                                  // 4 MB
    _Float16*  hT16 = (_Float16*)(ws + 4u * 1024 * 1024);          // 2 MB
    float*     s1   = (float*)(ws + 6u * 1024 * 1024);             // 64 KB
    float*     s2   = (float*)(ws + 6u * 1024 * 1024 + 64 * 1024); // 64 KB
    float*     M    = (float*)(ws + 6u * 1024 * 1024 + 128 * 1024);// 64 KB
    unsigned*  adjb = (unsigned*)(ws + 6u * 1024 * 1024 + 192 * 1024); // 512 KB

    gat_prep_kernel<<<2048, 256, 0, stream>>>(input, W, a, nw, nb, gam, bet,
                                              h32, hT16, s1, s2);
    gat_adjbits_kernel<<<512, 256, 0, stream>>>(adj, adjb);
    gat_rowmax_kernel<<<4096, 128, 0, stream>>>(adjb, s1, s2, M);
    gat_flash_kernel<<<256, 128, 0, stream>>>(hT16, h32, s1, s2, M, adjb, out);
}
